// HashRoutedNetwork_5557687681248
// MI455X (gfx1250) — matrix operand order repaired and checked by reading, not compiled, not run on hardware
//
#include <hip/hip_runtime.h>

typedef __bf16 bf16;
typedef __attribute__((ext_vector_type(16))) __bf16 bf16x16;
typedef __attribute__((ext_vector_type(8)))  __bf16 bf16x8;
typedef __attribute__((ext_vector_type(8)))  float  v8f;
typedef __attribute__((vector_size(16)))     int    i32x4;

#define B_N    4096
#define E_N    512
#define C_N    8
#define NUNITS 64
#define BK     16
#define DEPTH  8
#define EC     (E_N * C_N)      /* 4096  elems per sample        */
#define BC     (B_N * C_N)      /* 32768 GEMM columns            */
#define TOT    (B_N * E_N * C_N)/* 16777216 total elements       */
#define THR    1.0e-3f
#define EPSF   1.0e-8f

// ---------------------------------------------------------------------------
// WMMA fragment helpers (CDNA5 16-bit layouts, wave32)
// A 16x32: lane m=lane&15 is row; elems 0..7 = K[8*hi .. 8*hi+7],
//          elems 8..15 = K[8*hi+16 .. 8*hi+23]   (hi = lane>>4)
// B 32x16: lane n=lane&15 is col; same K order as A
// C/D 16x16 f32: row = v + 8*hi, col = lane&15
// ---------------------------------------------------------------------------
union FragU { bf16x16 v; bf16x8 h[2]; };
union Pack8 { bf16x8 v; bf16 e[8]; };

__device__ __forceinline__ bf16x16 ldA(const bf16* row, int kstep, int hi) {
    FragU f;
    const int kb = kstep * 32 + hi * 8;
    f.h[0] = *(const bf16x8*)(row + kb);
    f.h[1] = *(const bf16x8*)(row + kb + 16);
    return f.v;
}
__device__ __forceinline__ bf16x16 ldB(const bf16* col, int kstep, int hi) {
    FragU f;
    const int kb = kstep * 32 + hi * 8;
    f.h[0] = *(const bf16x8*)(col + kb);
    f.h[1] = *(const bf16x8*)(col + kb + 16);
    return f.v;
}

// ---------------------------------------------------------------------------
// 16 KB global->LDS panel stage (8192 bf16), 256 threads, no trailing barrier.
// Uses gfx1250 async-to-LDS path (ASYNCcnt) when the builtin is available.
// Builtin signature (from compiler diagnostic): param0 = i32x4* in AS(1).
// ---------------------------------------------------------------------------
#if defined(__has_builtin)
#if __has_builtin(__builtin_amdgcn_global_load_async_to_lds_b128) && \
    __has_builtin(__builtin_amdgcn_s_wait_asynccnt)
#define USE_ASYNC_LDS 1
#endif
#endif

__device__ __forceinline__ void stage16k(const bf16* __restrict__ g, bf16* s) {
    const int t = threadIdx.x;
#ifdef USE_ASYNC_LDS
    typedef __attribute__((address_space(1))) i32x4* gp_t;   // global src
    typedef __attribute__((address_space(3))) i32x4* lp_t;   // LDS dst
#pragma unroll
    for (int i = 0; i < 4; ++i) {
        const int elem = (i * 256 + t) * 8;               // 16 B per op
        gp_t gp = (gp_t)(unsigned long long)(const char*)(g + elem);
        lp_t lp = (lp_t)(unsigned int)(unsigned long long)(char*)(s + elem);
        __builtin_amdgcn_global_load_async_to_lds_b128(gp, lp, 0, 0);
    }
    __builtin_amdgcn_s_wait_asynccnt(0);
#else
#pragma unroll
    for (int i = 0; i < 4; ++i) {
        const int elem = (i * 256 + t) * 8;
        *(bf16x8*)(s + elem) = *(const bf16x8*)(g + elem);
    }
#endif
}

// ---------------------------------------------------------------------------
// One-time setup kernels
// ---------------------------------------------------------------------------
__global__ void k_init_r(const float* __restrict__ x, float* __restrict__ r_t) {
    int idx = blockIdx.x * blockDim.x + threadIdx.x;       // over (b,e,c)
    if (idx >= TOT) return;
    int b = idx >> 12;
    int rem = idx & 4095;
    int e = rem >> 3;
    int c = rem & 7;
    r_t[b * EC + c * E_N + e] = x[idx];
}

__global__ void k_cvt_hasher(const float* __restrict__ h, bf16* __restrict__ hb) {
    int idx = blockIdx.x * blockDim.x + threadIdx.x;
    if (idx < E_N * E_N) hb[idx] = (bf16)h[idx];
}

__global__ void k_cvt_basisT(const float* __restrict__ basis, bf16* __restrict__ bT) {
    int idx = blockIdx.x * blockDim.x + threadIdx.x;       // over (n,e,k)
    if (idx >= NUNITS * E_N * BK) return;
    int n = idx >> 13;
    int rem = idx & 8191;
    int e = rem >> 4;
    int k = rem & 15;
    bT[n * (E_N * BK) + k * E_N + e] = (bf16)basis[idx];
}

__global__ void k_zero_used(unsigned int* __restrict__ used32) {
    int idx = blockIdx.x * blockDim.x + threadIdx.x;
    if (idx < B_N * 2) used32[idx] = 0u;
}

// ---------------------------------------------------------------------------
// Per-depth: r -> bf16 copy + per-sample residual energy
// ---------------------------------------------------------------------------
__global__ void k_prep(const float* __restrict__ r_t, bf16* __restrict__ r_bf,
                       float* __restrict__ res_energy, float* __restrict__ init_energy,
                       int save_init) {
    __shared__ float red[256];
    const int b = blockIdx.x;
    const int t = threadIdx.x;
    const float* rp = r_t + (size_t)b * EC;
    bf16* wp = r_bf + (size_t)b * EC;
    float s = 0.f;
    for (int i = t; i < EC; i += 256) {
        float v = rp[i];
        wp[i] = (bf16)v;
        s += v * v;
    }
    red[t] = s;
    __syncthreads();
    for (int off = 128; off > 0; off >>= 1) {
        if (t < off) red[t] += red[t + off];
        __syncthreads();
    }
    if (t == 0) {
        res_energy[b] = red[0];
        if (save_init) init_energy[b] = red[0];
    }
}

// ---------------------------------------------------------------------------
// emb_t[(b,c)][e] = sum_f hasher[e][f] * r[f][(b,c)]   (bf16 WMMA, K=512)
// Block: 2 e-tiles (A panels in LDS) x 16 col tiles; wave: 2x2 tile block.
// ---------------------------------------------------------------------------
__global__ void __launch_bounds__(256) k_emb(const bf16* __restrict__ hasher_bf,
                                             const bf16* __restrict__ r_bf,
                                             bf16* __restrict__ emb_t) {
    __shared__ bf16 sA[2 * BK * E_N];                      // 2 x 16KB A panels
    const int lane = threadIdx.x & 31;
    const int wave = threadIdx.x >> 5;
    const int ep = blockIdx.x & 15;                        // e-tile pair (32 tiles)
    const int cg = blockIdx.x >> 4;                        // 128 column groups
    const int m = lane & 15;
    const int hi = lane >> 4;

    stage16k(hasher_bf + (size_t)(ep * 2 + 0) * 16 * E_N, sA);
    stage16k(hasher_bf + (size_t)(ep * 2 + 1) * 16 * E_N, sA + BK * E_N);
    __syncthreads();

    const int ct0 = cg * 16 + wave * 2;
    const int ct1 = ct0 + 1;
    const bf16* a0row = sA + (size_t)m * E_N;
    const bf16* a1row = sA + (size_t)(BK + m) * E_N;
    const bf16* b0col = r_bf + (size_t)(ct0 * 16 + m) * E_N;
    const bf16* b1col = r_bf + (size_t)(ct1 * 16 + m) * E_N;

    v8f acc00 = {}, acc01 = {}, acc10 = {}, acc11 = {};
#pragma unroll
    for (int ks = 0; ks < 16; ++ks) {
        bf16x16 a0 = ldA(a0row, ks, hi);
        bf16x16 a1 = ldA(a1row, ks, hi);
        bf16x16 b0 = ldB(b0col, ks, hi);
        bf16x16 b1 = ldB(b1col, ks, hi);
        acc00 = __builtin_amdgcn_wmma_f32_16x16x32_bf16(false, a0, false, b0, (short)0, acc00, false, false);
        acc01 = __builtin_amdgcn_wmma_f32_16x16x32_bf16(false, a0, false, b1, (short)0, acc01, false, false);
        acc10 = __builtin_amdgcn_wmma_f32_16x16x32_bf16(false, a1, false, b0, (short)0, acc10, false, false);
        acc11 = __builtin_amdgcn_wmma_f32_16x16x32_bf16(false, a1, false, b1, (short)0, acc11, false, false);
    }

    // D: row = v + 8*hi (=> e), col = lane&15 (=> bc); 8 consecutive e per lane
    const int e00 = (ep * 2 + 0) * 16 + hi * 8;
    const int e10 = (ep * 2 + 1) * 16 + hi * 8;
    const int bc0 = ct0 * 16 + m;
    const int bc1 = ct1 * 16 + m;
    Pack8 p;
#pragma unroll
    for (int v = 0; v < 8; ++v) p.e[v] = (bf16)acc00[v];
    *(bf16x8*)(emb_t + (size_t)bc0 * E_N + e00) = p.v;
#pragma unroll
    for (int v = 0; v < 8; ++v) p.e[v] = (bf16)acc01[v];
    *(bf16x8*)(emb_t + (size_t)bc1 * E_N + e00) = p.v;
#pragma unroll
    for (int v = 0; v < 8; ++v) p.e[v] = (bf16)acc10[v];
    *(bf16x8*)(emb_t + (size_t)bc0 * E_N + e10) = p.v;
#pragma unroll
    for (int v = 0; v < 8; ++v) p.e[v] = (bf16)acc11[v];
    *(bf16x8*)(emb_t + (size_t)bc1 * E_N + e10) = p.v;
}

// ---------------------------------------------------------------------------
// inv_norm[bc] = 1 / (sqrt(sum_e emb^2) + EPS)
// ---------------------------------------------------------------------------
__global__ void k_norm(const bf16* __restrict__ emb_t, float* __restrict__ inv_norm) {
    const int lane = threadIdx.x & 31;
    const int wave = threadIdx.x >> 5;
    const int col = blockIdx.x * 8 + wave;
    const bf16* cp = emb_t + (size_t)col * E_N + lane * 16;
    float s = 0.f;
#pragma unroll
    for (int i = 0; i < 16; ++i) {
        float v = (float)cp[i];
        s += v * v;
    }
#pragma unroll
    for (int off = 16; off > 0; off >>= 1) s += __shfl_xor(s, off, 32);
    if (lane == 0) inv_norm[col] = 1.f / (sqrtf(s) + EPSF);
}

// ---------------------------------------------------------------------------
// scores[b][n] = sum_c inv_norm[b,c]^2 * sum_k (basis[n]^T emb[b])_{k,c}^2
// Block: 2 units (A panels in LDS) x 16 col tiles; wave: 2x2 tile block.
// ---------------------------------------------------------------------------
__device__ __forceinline__ void score_store(v8f acc, int coltile, int n,
                                            const float* __restrict__ inv_norm,
                                            float* __restrict__ scores, int lane) {
    float sk = 0.f;
#pragma unroll
    for (int v = 0; v < 8; ++v) sk += acc[v] * acc[v];
    sk += __shfl_xor(sk, 16, 32);
    const float inv = inv_norm[coltile * 16 + (lane & 15)];
    float val = sk * inv * inv;
    val += __shfl_xor(val, 1, 32);
    val += __shfl_xor(val, 2, 32);
    val += __shfl_xor(val, 4, 32);
    if (lane == 0)      scores[(size_t)(coltile * 2 + 0) * NUNITS + n] = val;
    else if (lane == 8) scores[(size_t)(coltile * 2 + 1) * NUNITS + n] = val;
}

__global__ void __launch_bounds__(256) k_scores(const bf16* __restrict__ basisT,
                                                const bf16* __restrict__ emb_t,
                                                const float* __restrict__ inv_norm,
                                                float* __restrict__ scores) {
    __shared__ bf16 sA[2 * BK * E_N];                      // 2 x 16KB unit panels
    const int lane = threadIdx.x & 31;
    const int wave = threadIdx.x >> 5;
    const int np = blockIdx.x & 31;                        // unit pair (64 units)
    const int cg = blockIdx.x >> 5;                        // 128 column groups
    const int m = lane & 15;
    const int hi = lane >> 4;
    const int n0 = np * 2, n1 = np * 2 + 1;

    stage16k(basisT + (size_t)n0 * (BK * E_N), sA);
    stage16k(basisT + (size_t)n1 * (BK * E_N), sA + BK * E_N);
    __syncthreads();

    const int ct0 = cg * 16 + wave * 2;
    const int ct1 = ct0 + 1;
    const bf16* a0row = sA + (size_t)m * E_N;
    const bf16* a1row = sA + (size_t)(BK + m) * E_N;
    const bf16* b0col = emb_t + (size_t)(ct0 * 16 + m) * E_N;
    const bf16* b1col = emb_t + (size_t)(ct1 * 16 + m) * E_N;

    v8f acc00 = {}, acc01 = {}, acc10 = {}, acc11 = {};
#pragma unroll
    for (int ks = 0; ks < 16; ++ks) {
        bf16x16 a0 = ldA(a0row, ks, hi);
        bf16x16 a1 = ldA(a1row, ks, hi);
        bf16x16 b0 = ldB(b0col, ks, hi);
        bf16x16 b1 = ldB(b1col, ks, hi);
        acc00 = __builtin_amdgcn_wmma_f32_16x16x32_bf16(false, a0, false, b0, (short)0, acc00, false, false);
        acc01 = __builtin_amdgcn_wmma_f32_16x16x32_bf16(false, a0, false, b1, (short)0, acc01, false, false);
        acc10 = __builtin_amdgcn_wmma_f32_16x16x32_bf16(false, a1, false, b0, (short)0, acc10, false, false);
        acc11 = __builtin_amdgcn_wmma_f32_16x16x32_bf16(false, a1, false, b1, (short)0, acc11, false, false);
    }

    score_store(acc00, ct0, n0, inv_norm, scores, lane);
    score_store(acc01, ct1, n0, inv_norm, scores, lane);
    score_store(acc10, ct0, n1, inv_norm, scores, lane);
    score_store(acc11, ct1, n1, inv_norm, scores, lane);
}

// ---------------------------------------------------------------------------
// Routing: masked argmax over 64 units + active gate + used update
// ---------------------------------------------------------------------------
__global__ void k_route(const float* __restrict__ scores, unsigned long long* __restrict__ used,
                        const float* __restrict__ res_energy, const float* __restrict__ init_energy,
                        int* __restrict__ best, int* __restrict__ actv) {
    const int lane = threadIdx.x & 31;
    const int wave = threadIdx.x >> 5;
    const int b = blockIdx.x * 8 + wave;
    const unsigned long long u = used[b];
    const float NEG = -3.0e38f;

    int i1 = lane, i2 = lane + 32;
    float s1 = ((u >> i1) & 1ull) ? NEG : scores[(size_t)b * NUNITS + i1];
    float s2 = ((u >> i2) & 1ull) ? NEG : scores[(size_t)b * NUNITS + i2];
    float v = s1; int i = i1;
    if (s2 > v) { v = s2; i = i2; }
#pragma unroll
    for (int off = 16; off > 0; off >>= 1) {
        float ov = __shfl_xor(v, off, 32);
        int   oi = __shfl_xor(i, off, 32);
        if (ov > v || (ov == v && oi < i)) { v = ov; i = oi; }
    }
    if (lane == 0) {
        int active = res_energy[b] > THR * init_energy[b];
        best[b] = i;
        actv[b] = active;
        if (active) used[b] = u | (1ull << i);
    }
}

// ---------------------------------------------------------------------------
// Projection + residual update (fp32 exact path):
// coeffs = b_sel^T r[b];  r[b] -= b_sel coeffs   (only if active)
// ---------------------------------------------------------------------------
__global__ void k_proj(const float* __restrict__ basis, const int* __restrict__ best,
                       const int* __restrict__ actv, float* __restrict__ r_t) {
    const int b = blockIdx.x;
    if (!actv[b]) return;                      // block-uniform exit
    const int t = threadIdx.x;
    const int n = best[b];

    __shared__ float sb[BK * E_N];             // b_sel transposed [k][e], 32 KB
    __shared__ float sc[BK * C_N];             // coeffs [k][c]

    const float* bp = basis + (size_t)n * (E_N * BK);
    for (int idx = t; idx < E_N * BK; idx += 256) {
        int e = idx >> 4;
        int k = idx & 15;
        sb[k * E_N + e] = bp[idx];
    }
    __syncthreads();

    float* rp = r_t + (size_t)b * EC;
    if (t < BK * C_N) {
        int k = t >> 3, c = t & 7;
        const float* rc = rp + c * E_N;
        const float* bk = sb + k * E_N;
        float dot = 0.f;
        for (int e = 0; e < E_N; ++e) dot = fmaf(bk[e], rc[e], dot);
        sc[k * C_N + c] = dot;
    }
    __syncthreads();

    for (int idx = t; idx < EC; idx += 256) {  // idx = c*512 + e
        int c = idx >> 9;
        int e = idx & 511;
        float p = 0.f;
#pragma unroll
        for (int k = 0; k < BK; ++k) p = fmaf(sb[k * E_N + e], sc[k * C_N + c], p);
        rp[idx] -= p;
    }
}

// ---------------------------------------------------------------------------
// out = x - r_final   (exact identity: out accumulates every subtracted proj)
// ---------------------------------------------------------------------------
__global__ void k_out(const float* __restrict__ x, const float* __restrict__ r_t,
                      float* __restrict__ out) {
    int idx = blockIdx.x * blockDim.x + threadIdx.x;
    if (idx >= TOT) return;
    int b = idx >> 12;
    int rem = idx & 4095;
    int e = rem >> 3;
    int c = rem & 7;
    out[idx] = x[idx] - r_t[(size_t)b * EC + c * E_N + e];
}

// ---------------------------------------------------------------------------
extern "C" void kernel_launch(void* const* d_in, const int* in_sizes, int n_in,
                              void* d_out, int out_size, void* d_ws, size_t ws_size,
                              hipStream_t stream) {
    (void)in_sizes; (void)n_in; (void)out_size; (void)ws_size;
    const float* x      = (const float*)d_in[0];
    const float* basis  = (const float*)d_in[1];
    const float* hasher = (const float*)d_in[2];
    float* out = (float*)d_out;

    char* ws = (char*)d_ws;
    size_t off = 0;
    float* r_t        = (float*)(ws + off); off += (size_t)TOT * 4;        // 64 MB
    bf16*  r_bf       = (bf16*) (ws + off); off += (size_t)TOT * 2;        // 32 MB
    bf16*  emb_t      = (bf16*) (ws + off); off += (size_t)TOT * 2;        // 32 MB
    bf16*  hasher_bf  = (bf16*) (ws + off); off += (size_t)E_N * E_N * 2;  // 512 KB
    bf16*  basisT     = (bf16*) (ws + off); off += (size_t)NUNITS * E_N * BK * 2; // 1 MB
    float* inv_norm   = (float*)(ws + off); off += (size_t)BC * 4;         // 128 KB
    float* res_energy = (float*)(ws + off); off += (size_t)B_N * 4;
    float* init_energy= (float*)(ws + off); off += (size_t)B_N * 4;
    float* scores     = (float*)(ws + off); off += (size_t)B_N * NUNITS * 4; // 1 MB
    int*   best       = (int*)  (ws + off); off += (size_t)B_N * 4;
    int*   actv       = (int*)  (ws + off); off += (size_t)B_N * 4;
    unsigned long long* used = (unsigned long long*)(ws + off); off += (size_t)B_N * 8;

    // one-time setup
    k_init_r<<<TOT / 256, 256, 0, stream>>>(x, r_t);
    k_cvt_hasher<<<(E_N * E_N) / 256, 256, 0, stream>>>(hasher, hasher_bf);
    k_cvt_basisT<<<(NUNITS * E_N * BK) / 256, 256, 0, stream>>>(basis, basisT);
    k_zero_used<<<(B_N * 2) / 256, 256, 0, stream>>>((unsigned int*)used);

    for (int d = 0; d < DEPTH; ++d) {
        k_prep<<<B_N, 256, 0, stream>>>(r_t, r_bf, res_energy, init_energy, d == 0);
        k_emb<<<(16) * 128, 256, 0, stream>>>(hasher_bf, r_bf, emb_t);          // 2048 blocks
        k_norm<<<BC / 8, 256, 0, stream>>>(emb_t, inv_norm);
        k_scores<<<(32) * 128, 256, 0, stream>>>(basisT, emb_t, inv_norm, scores); // 4096 blocks
        k_route<<<B_N / 8, 256, 0, stream>>>(scores, used, res_energy, init_energy, best, actv);
        k_proj<<<B_N, 256, 0, stream>>>(basis, best, actv, r_t);
    }

    k_out<<<TOT / 256, 256, 0, stream>>>(x, r_t, out);
}
